// MultimodalTransformerMultiToken_79182017069620
// MI455X (gfx1250) — compile-verified, matrix operand order checked by reading, not run on hardware
//
#include <hip/hip_runtime.h>
#include <hip/hip_bf16.h>

typedef __bf16 v16bf __attribute__((ext_vector_type(16)));
typedef float  v8f   __attribute__((ext_vector_type(8)));

#define DEV __device__ __forceinline__

DEV float bfu2f(unsigned short u) {
    unsigned int x = ((unsigned int)u) << 16;
    return __uint_as_float(x);
}
DEV unsigned short f2bfu(float f) {
    __bf16 h = (__bf16)f;
    return __builtin_bit_cast(unsigned short, h);
}
DEV float gelu_exact(float x) {
    return 0.5f * x * (1.f + erff(x * 0.70710678118654752f));
}

// ---------------------------------------------------------------------------
// WMMA fragment layout helpers (CDNA5 ISA 7.12.2, 16-bit 16x32 A / 32x16 B):
//   lane l: m/n = l&15, K-half = l>>4 ; elem e: v=e>>1, g=v>>2, kp=v&3
//   k_local = g*16 + khalf*8 + kp*2 + (e&1)
// Fragments stored in LDS pre-swizzled: frag f occupies 512 __bf16 at
// f*512 + lane*16 + e so each lane loads one contiguous 32-byte v16bf.
// ---------------------------------------------------------------------------
DEV int frag_k_local(int l, int e) {
    int kh = l >> 4;
    int v = e >> 1, g = v >> 2, kp = v & 3;
    return g * 16 + kh * 8 + kp * 2 + (e & 1);
}

// offset of element (m, k) inside an A-fragment chain (kt = k>>5)
DEV int afrag_off(int m, int k) {
    int kt = k >> 5, kl = k & 31;
    int lsb = kl & 1, t2 = kl >> 1;
    int g = t2 >> 3, kh = (t2 >> 2) & 1, kp = t2 & 3;
    int v = g * 4 + kp, e = v * 2 + lsb;
    int l = kh * 16 + m;
    return kt * 512 + l * 16 + e;
}

DEV v16bf ld_frag(const __bf16* base, int frag_idx, int lane) {
    return *(const v16bf*)(base + frag_idx * 512 + lane * 16);
}

DEV v8f wmma_bf16(v16bf a, v16bf b, v8f c) {
    return __builtin_amdgcn_wmma_f32_16x16x32_bf16(false, a, false, b,
                                                   (short)0, c, false, false);
}

// Block-cooperative: load W (f32, row-major, leading dim ld) into pre-swizzled
// B-fragments. KT = K/32, NT = N-tiles, cols >= nvalid are zero padded.
DEV void load_w_frags(__bf16* dst, const float* __restrict__ src, int ld,
                      int KT, int NT, int nvalid, int tid, int nthreads) {
    int total = KT * NT * 512;
    for (int s = tid; s < total; s += nthreads) {
        int e = s & 15;
        int l = (s >> 4) & 31;
        int t = s >> 9;
        int nt = t % NT, kt = t / NT;
        int n = nt * 16 + (l & 15);
        int k = kt * 32 + frag_k_local(l, e);
        float f = (n < nvalid) ? src[k * ld + n] : 0.f;
        dst[s] = (__bf16)f;
    }
}

// Per-wave: load 16 rows of A (f32, leading dim ld) into A-fragments.
DEV void load_a_frags_f32(__bf16* dst, const float* __restrict__ src, int ld,
                          int KT, int lane) {
    int total = KT * 512;
    for (int s = lane; s < total; s += 32) {
        int e = s & 15, l = (s >> 4) & 31, kt = s >> 9;
        int m = l & 15;
        int k = kt * 32 + frag_k_local(l, e);
        dst[s] = (__bf16)src[m * ld + k];
    }
}
DEV void load_a_frags_bf16(__bf16* dst, const unsigned short* __restrict__ src,
                           int ld, int KT, int lane) {
    int total = KT * 512;
    for (int s = lane; s < total; s += 32) {
        int e = s & 15, l = (s >> 4) & 31, kt = s >> 9;
        int m = l & 15;
        int k = kt * 32 + frag_k_local(l, e);
        dst[s] = __builtin_bit_cast(__bf16, src[m * ld + k]);
    }
}

// LayerNorm over 8 N-tiles (128 cols) held as C-fragments, then store f32.
// C layout: lane l holds n = nt*16 + (l&15); VGPR v holds m = v + 8*(l>>4).
DEV void ln_store(v8f* y, int lane, int r0, const float* g, const float* b,
                  float* __restrict__ dst, float* __restrict__ dst2) {
    int half = lane >> 4, nl = lane & 15;
#pragma unroll
    for (int v = 0; v < 8; ++v) {
        int m = v + 8 * half;
        float s = 0.f;
#pragma unroll
        for (int nt = 0; nt < 8; ++nt) s += y[nt][v];
#pragma unroll
        for (int mk = 1; mk <= 8; mk <<= 1) s += __shfl_xor(s, mk, 32);
        float mu = s * (1.f / 128.f);
        float q = 0.f;
#pragma unroll
        for (int nt = 0; nt < 8; ++nt) {
            float d = y[nt][v] - mu;
            q += d * d;
        }
#pragma unroll
        for (int mk = 1; mk <= 8; mk <<= 1) q += __shfl_xor(q, mk, 32);
        float rstd = rsqrtf(q * (1.f / 128.f) + 1e-5f);
        long long row = r0 + m;
#pragma unroll
        for (int nt = 0; nt < 8; ++nt) {
            int n = nt * 16 + nl;
            float o = (y[nt][v] - mu) * rstd * g[n] + b[n];
            dst[row * 128 + n] = o;
            if (dst2) dst2[row * 128 + n] = o;
        }
    }
}

// ---------------------------------------------------------------------------
// Fused modality encoder: h = x@pW + pb + pos ; y = LN(h + MLP(h))
// 2 waves/block, wave owns 16 rows of (B*T) x full D=128.
// ---------------------------------------------------------------------------
template <int T, int R>
__global__ __launch_bounds__(64) void encoder_kernel(
    const float* __restrict__ x, const float* __restrict__ pW,
    const float* __restrict__ pb, const float* __restrict__ pos,
    const float* __restrict__ lng, const float* __restrict__ lnb,
    const float* __restrict__ W1, const float* __restrict__ b1,
    const float* __restrict__ W2, const float* __restrict__ b2,
    float* __restrict__ Eout) {
    constexpr int KTP = R / 32;
    __shared__ __bf16 w_frag[4 * 8 * 512];
    __shared__ __bf16 a_frag[2][4 * 512];
    __shared__ __bf16 g_frag[2][4 * 512];
    __shared__ float s_pb[128], s_pos[T * 128], s_b1[256], s_b2[128];
    __shared__ float s_g[128], s_b[128];

    int tid = threadIdx.x, lane = tid & 31, wid = tid >> 5;
    int r0 = blockIdx.x * 32 + wid * 16;

    for (int i = tid; i < 128; i += 64) {
        s_pb[i] = pb[i]; s_b2[i] = b2[i]; s_g[i] = lng[i]; s_b[i] = lnb[i];
    }
    for (int i = tid; i < 256; i += 64) s_b1[i] = b1[i];
    for (int i = tid; i < T * 128; i += 64) s_pos[i] = pos[i];

    load_w_frags(w_frag, pW, 128, KTP, 8, 128, tid, 64);
    load_a_frags_f32(a_frag[wid], x + (long long)r0 * R, R, KTP, lane);
    __syncthreads();

    v8f zero = {};
    v8f h[8];
#pragma unroll
    for (int nt = 0; nt < 8; ++nt) h[nt] = zero;
#pragma unroll
    for (int kt = 0; kt < KTP; ++kt) {
        v16bf a = ld_frag(a_frag[wid], kt, lane);
#pragma unroll
        for (int nt = 0; nt < 8; ++nt)
            h[nt] = wmma_bf16(a, ld_frag(w_frag, kt * 8 + nt, lane), h[nt]);
    }

    int nl = lane & 15, half = lane >> 4;
#pragma unroll
    for (int nt = 0; nt < 8; ++nt) {
        int n = nt * 16 + nl;
#pragma unroll
        for (int v = 0; v < 8; ++v) {
            int m = v + 8 * half;
            int row = r0 + m;
            int t = row % T;
            float f = h[nt][v] + s_pb[n] + s_pos[t * 128 + n];
            h[nt][v] = f;                         // residual, kept in regs
            a_frag[wid][afrag_off(m, n)] = (__bf16)f;  // A operand for MLP
        }
    }

    v8f mlp[8];
#pragma unroll
    for (int nt = 0; nt < 8; ++nt) mlp[nt] = zero;

    for (int hh = 0; hh < 2; ++hh) {  // MLP1 N-half == MLP2 K-half
        __syncthreads();
        load_w_frags(w_frag, W1 + hh * 128, 256, 4, 8, 128, tid, 64);
        __syncthreads();
        v8f o1[8];
#pragma unroll
        for (int nt = 0; nt < 8; ++nt) o1[nt] = zero;
#pragma unroll
        for (int kt = 0; kt < 4; ++kt) {
            v16bf a = ld_frag(a_frag[wid], kt, lane);
#pragma unroll
            for (int nt = 0; nt < 8; ++nt)
                o1[nt] = wmma_bf16(a, ld_frag(w_frag, kt * 8 + nt, lane), o1[nt]);
        }
#pragma unroll
        for (int nt = 0; nt < 8; ++nt) {
#pragma unroll
            for (int v = 0; v < 8; ++v) {
                float f = gelu_exact(o1[nt][v] + s_b1[hh * 128 + nt * 16 + nl]);
                g_frag[wid][afrag_off(v + 8 * half, nt * 16 + nl)] = (__bf16)f;
            }
        }
        __syncthreads();
        load_w_frags(w_frag, W2 + hh * 128 * 128, 128, 4, 8, 128, tid, 64);
        __syncthreads();
#pragma unroll
        for (int kt = 0; kt < 4; ++kt) {
            v16bf a = ld_frag(g_frag[wid], kt, lane);
#pragma unroll
            for (int nt = 0; nt < 8; ++nt)
                mlp[nt] = wmma_bf16(a, ld_frag(w_frag, kt * 8 + nt, lane), mlp[nt]);
        }
    }

#pragma unroll
    for (int nt = 0; nt < 8; ++nt)
#pragma unroll
        for (int v = 0; v < 8; ++v)
            h[nt][v] = h[nt][v] + mlp[nt][v] + s_b2[nt * 16 + nl];

    ln_store(h, lane, r0, s_g, s_b, Eout, nullptr);
}

// ---------------------------------------------------------------------------
// Generic 128x128 projection GEMM: out_bf16 = A_f32 @ W + bias
// ---------------------------------------------------------------------------
__global__ __launch_bounds__(64) void proj_gemm_kernel(
    const float* __restrict__ A, const float* __restrict__ W,
    const float* __restrict__ bias, unsigned short* __restrict__ out) {
    __shared__ __bf16 w_frag[4 * 8 * 512];
    __shared__ __bf16 a_frag[2][4 * 512];
    __shared__ float s_b[128];

    int tid = threadIdx.x, lane = tid & 31, wid = tid >> 5;
    int r0 = blockIdx.x * 32 + wid * 16;

    for (int i = tid; i < 128; i += 64) s_b[i] = bias[i];
    load_w_frags(w_frag, W, 128, 4, 8, 128, tid, 64);
    load_a_frags_f32(a_frag[wid], A + (long long)r0 * 128, 128, 4, lane);
    __syncthreads();

    v8f zero = {};
    v8f c[8];
#pragma unroll
    for (int nt = 0; nt < 8; ++nt) c[nt] = zero;
#pragma unroll
    for (int kt = 0; kt < 4; ++kt) {
        v16bf a = ld_frag(a_frag[wid], kt, lane);
#pragma unroll
        for (int nt = 0; nt < 8; ++nt)
            c[nt] = wmma_bf16(a, ld_frag(w_frag, kt * 8 + nt, lane), c[nt]);
    }
    int nl = lane & 15, half = lane >> 4;
#pragma unroll
    for (int nt = 0; nt < 8; ++nt) {
        int n = nt * 16 + nl;
#pragma unroll
        for (int v = 0; v < 8; ++v) {
            long long row = r0 + v + 8 * half;
            out[row * 128 + n] = f2bfu(c[nt][v] + s_b[n]);
        }
    }
}

// ---------------------------------------------------------------------------
// O-projection + residual + LayerNorm: dst = LN(resid + attn@W + b)
// ---------------------------------------------------------------------------
__global__ __launch_bounds__(64) void oproj_ln_kernel(
    const unsigned short* __restrict__ Attn, const float* __restrict__ resid,
    const float* __restrict__ W, const float* __restrict__ bias,
    const float* __restrict__ lng, const float* __restrict__ lnb,
    float* __restrict__ dst, float* __restrict__ dst2) {
    __shared__ __bf16 w_frag[4 * 8 * 512];
    __shared__ __bf16 a_frag[2][4 * 512];
    __shared__ float s_b[128], s_g[128], s_bb[128];

    int tid = threadIdx.x, lane = tid & 31, wid = tid >> 5;
    int r0 = blockIdx.x * 32 + wid * 16;

    for (int i = tid; i < 128; i += 64) {
        s_b[i] = bias[i]; s_g[i] = lng[i]; s_bb[i] = lnb[i];
    }
    load_w_frags(w_frag, W, 128, 4, 8, 128, tid, 64);
    load_a_frags_bf16(a_frag[wid], Attn + (long long)r0 * 128, 128, 4, lane);
    __syncthreads();

    v8f zero = {};
    v8f c[8];
#pragma unroll
    for (int nt = 0; nt < 8; ++nt) c[nt] = zero;
#pragma unroll
    for (int kt = 0; kt < 4; ++kt) {
        v16bf a = ld_frag(a_frag[wid], kt, lane);
#pragma unroll
        for (int nt = 0; nt < 8; ++nt)
            c[nt] = wmma_bf16(a, ld_frag(w_frag, kt * 8 + nt, lane), c[nt]);
    }
    int nl = lane & 15, half = lane >> 4;
#pragma unroll
    for (int nt = 0; nt < 8; ++nt) {
        int n = nt * 16 + nl;
#pragma unroll
        for (int v = 0; v < 8; ++v) {
            long long row = r0 + v + 8 * half;
            c[nt][v] += s_b[n] + resid[row * 128 + n];
        }
    }
    ln_store(c, lane, r0, s_g, s_bb, dst, dst2);
}

// ---------------------------------------------------------------------------
// Attention core: one thread per (batch, head). Tq,Tk <= 5 so everything
// lives in registers; head-average of softmax weights via shfl_xor over the
// 4 adjacent head lanes.
// ---------------------------------------------------------------------------
template <int Tq, int Tk>
__global__ __launch_bounds__(256) void attn_core_kernel(
    const unsigned short* __restrict__ Q, const unsigned short* __restrict__ K,
    const unsigned short* __restrict__ V, unsigned short* __restrict__ attn_out,
    float* __restrict__ wavg) {
    int idx = blockIdx.x * 256 + threadIdx.x;
    long long b = idx >> 2;
    int h = idx & 3;
    const unsigned short* Qb = Q + (b * Tq) * 128 + h * 32;
    const unsigned short* Kb = K + (b * Tk) * 128 + h * 32;
    const unsigned short* Vb = V + (b * Tk) * 128 + h * 32;

    float s[Tq][Tk];
#pragma unroll
    for (int q = 0; q < Tq; ++q) {
        float qr[32];
#pragma unroll
        for (int d = 0; d < 32; ++d) qr[d] = bfu2f(Qb[q * 128 + d]);
#pragma unroll
        for (int k = 0; k < Tk; ++k) {
            float acc = 0.f;
#pragma unroll
            for (int d = 0; d < 32; ++d) acc += qr[d] * bfu2f(Kb[k * 128 + d]);
            s[q][k] = acc * 0.17677669529663687f;  // 1/sqrt(32)
        }
    }
#pragma unroll
    for (int q = 0; q < Tq; ++q) {
        float mx = s[q][0];
#pragma unroll
        for (int k = 1; k < Tk; ++k) mx = fmaxf(mx, s[q][k]);
        float sum = 0.f;
#pragma unroll
        for (int k = 0; k < Tk; ++k) {
            s[q][k] = expf(s[q][k] - mx);
            sum += s[q][k];
        }
        float inv = 1.f / sum;
#pragma unroll
        for (int k = 0; k < Tk; ++k) s[q][k] *= inv;
    }
    // head-averaged weights (heads occupy 4 consecutive lanes)
#pragma unroll
    for (int q = 0; q < Tq; ++q)
#pragma unroll
        for (int k = 0; k < Tk; ++k) {
            float v = s[q][k];
            v += __shfl_xor(v, 1, 32);
            v += __shfl_xor(v, 2, 32);
            if (h == 0) wavg[b * (Tq * Tk) + q * Tk + k] = v * 0.25f;
        }
    // attn = weights @ V
#pragma unroll
    for (int q = 0; q < Tq; ++q) {
        float acc[32];
#pragma unroll
        for (int d = 0; d < 32; ++d) acc[d] = 0.f;
#pragma unroll
        for (int k = 0; k < Tk; ++k) {
            float w = s[q][k];
#pragma unroll
            for (int d = 0; d < 32; ++d) acc[d] += w * bfu2f(Vb[k * 128 + d]);
        }
#pragma unroll
        for (int d = 0; d < 32; ++d)
            attn_out[(b * Tq + q) * 128 + h * 32 + d] = f2bfu(acc[d]);
    }
}

// ---------------------------------------------------------------------------
// Classifier: pooled(B,384) -> W1 -> GELU -> W2 -> logits(B,200)
// K-chunks of W1 correspond exactly to the 3 modalities (D=128 each).
// ---------------------------------------------------------------------------
__global__ __launch_bounds__(64) void classifier_kernel(
    const float* __restrict__ Ea, const float* __restrict__ Et,
    const float* __restrict__ Ev, const float* __restrict__ W1,
    const float* __restrict__ b1, const float* __restrict__ W2,
    const float* __restrict__ b2, float* __restrict__ logits) {
    __shared__ __bf16 w_frag[4 * 8 * 512];
    __shared__ __bf16 a_frag[2][4 * 512];
    __shared__ __bf16 g_frag[2][4 * 512];
    __shared__ float s_b1[128], s_b2[200];

    int tid = threadIdx.x, lane = tid & 31, wid = tid >> 5;
    int r0 = blockIdx.x * 32 + wid * 16;

    for (int i = tid; i < 128; i += 64) s_b1[i] = b1[i];
    for (int i = tid; i < 200; i += 64) s_b2[i] = b2[i];

    v8f zero = {};
    v8f hid[8];
#pragma unroll
    for (int nt = 0; nt < 8; ++nt) hid[nt] = zero;

    const float* Es[3] = {Ea, Et, Ev};
    const int Ts[3] = {5, 3, 4};
    for (int c = 0; c < 3; ++c) {
        __syncthreads();
        load_w_frags(w_frag, W1 + c * 128 * 128, 128, 4, 8, 128, tid, 64);
        const float* E = Es[c];
        int T = Ts[c];
        float inv = 1.f / (float)T;
        for (int s2 = lane; s2 < 2048; s2 += 32) {  // pooled tile (16x128)
            int e = s2 & 15, l = (s2 >> 4) & 31, kt = s2 >> 9;
            int m = l & 15;
            int k = kt * 32 + frag_k_local(l, e);
            float acc = 0.f;
            for (int t = 0; t < T; ++t)
                acc += E[((long long)(r0 + m) * T + t) * 128 + k];
            a_frag[wid][s2] = (__bf16)(acc * inv);
        }
        __syncthreads();
#pragma unroll
        for (int kt = 0; kt < 4; ++kt) {
            v16bf a = ld_frag(a_frag[wid], kt, lane);
#pragma unroll
            for (int nt = 0; nt < 8; ++nt)
                hid[nt] = wmma_bf16(a, ld_frag(w_frag, kt * 8 + nt, lane), hid[nt]);
        }
    }
    int nl = lane & 15, half = lane >> 4;
#pragma unroll
    for (int nt = 0; nt < 8; ++nt)
#pragma unroll
        for (int v = 0; v < 8; ++v) {
            float f = gelu_exact(hid[nt][v] + s_b1[nt * 16 + nl]);
            g_frag[wid][afrag_off(v + 8 * half, nt * 16 + nl)] = (__bf16)f;
        }
    // logits cols 0..127
    __syncthreads();
    load_w_frags(w_frag, W2, 200, 4, 8, 128, tid, 64);
    __syncthreads();
    v8f lg[8];
#pragma unroll
    for (int nt = 0; nt < 8; ++nt) lg[nt] = zero;
#pragma unroll
    for (int kt = 0; kt < 4; ++kt) {
        v16bf a = ld_frag(g_frag[wid], kt, lane);
#pragma unroll
        for (int nt = 0; nt < 8; ++nt)
            lg[nt] = wmma_bf16(a, ld_frag(w_frag, kt * 8 + nt, lane), lg[nt]);
    }
#pragma unroll
    for (int nt = 0; nt < 8; ++nt)
#pragma unroll
        for (int v = 0; v < 8; ++v) {
            long long row = r0 + v + 8 * half;
            int n = nt * 16 + nl;
            logits[row * 200 + n] = lg[nt][v] + s_b2[n];
        }
    // logits cols 128..199 (5 tiles, 72 valid cols)
    __syncthreads();
    load_w_frags(w_frag, W2 + 128, 200, 4, 5, 72, tid, 64);
    __syncthreads();
    v8f lg2[5];
#pragma unroll
    for (int nt = 0; nt < 5; ++nt) lg2[nt] = zero;
#pragma unroll
    for (int kt = 0; kt < 4; ++kt) {
        v16bf a = ld_frag(g_frag[wid], kt, lane);
#pragma unroll
        for (int nt = 0; nt < 5; ++nt)
            lg2[nt] = wmma_bf16(a, ld_frag(w_frag, kt * 5 + nt, lane), lg2[nt]);
    }
#pragma unroll
    for (int nt = 0; nt < 5; ++nt)
#pragma unroll
        for (int v = 0; v < 8; ++v) {
            long long row = r0 + v + 8 * half;
            int n = 128 + nt * 16 + nl;
            if (n < 200) logits[row * 200 + n] = lg2[nt][v] + s_b2[n];
        }
}

// ---------------------------------------------------------------------------
extern "C" void kernel_launch(void* const* d_in, const int* in_sizes, int n_in,
                              void* d_out, int out_size, void* d_ws,
                              size_t ws_size, hipStream_t stream) {
    (void)in_sizes; (void)n_in; (void)out_size; (void)ws_size;
    constexpr long long B = 32768;

    const float* x_a = (const float*)d_in[0];
    const float* x_t = (const float*)d_in[1];
    const float* x_v = (const float*)d_in[2];
    auto P = [&](int i) { return (const float*)d_in[i]; };
    float* out = (float*)d_out;

    // workspace layout
    char* ws = (char*)d_ws;
    size_t off = 0;
    float* Ea = (float*)(ws + off); off += (size_t)B * 5 * 128 * 4;
    float* Et = (float*)(ws + off); off += (size_t)B * 3 * 128 * 4;
    float* Ev = (float*)(ws + off); off += (size_t)B * 4 * 128 * 4;
    unsigned short* Qb = (unsigned short*)(ws + off); off += (size_t)B * 5 * 128 * 2;
    unsigned short* Kb = (unsigned short*)(ws + off); off += (size_t)B * 5 * 128 * 2;
    unsigned short* Vb = (unsigned short*)(ws + off); off += (size_t)B * 5 * 128 * 2;
    unsigned short* Ab = (unsigned short*)(ws + off); off += (size_t)B * 5 * 128 * 2;

    // output layout: logits, 6 attn averages, 3 encoded (reference order)
    size_t oo = 0;
    float* o_logits = out; oo += (size_t)B * 200;
    const int wsz[6] = {15, 20, 15, 12, 20, 12};
    float* o_w[6];
    for (int i = 0; i < 6; ++i) { o_w[i] = out + oo; oo += (size_t)B * wsz[i]; }
    float* o_Ea = out + oo; oo += (size_t)B * 640;
    float* o_Et = out + oo; oo += (size_t)B * 384;
    float* o_Ev = out + oo;

    // ---- encoders ----
    encoder_kernel<5, 64><<<(int)(B * 5 / 32), 64, 0, stream>>>(
        x_a, P(3), P(4), P(5), P(6), P(7), P(8), P(9), P(10), P(11), Ea);
    encoder_kernel<3, 32><<<(int)(B * 3 / 32), 64, 0, stream>>>(
        x_t, P(12), P(13), P(14), P(15), P(16), P(17), P(18), P(19), P(20), Et);
    encoder_kernel<4, 128><<<(int)(B * 4 / 32), 64, 0, stream>>>(
        x_v, P(21), P(22), P(23), P(24), P(25), P(26), P(27), P(28), P(29), Ev);

    // ---- sequential cross-attention pairs ----
    struct PairDesc { int Tq, Tk, pb; float *Eq, *Ekv, *wout, *enc; };
    PairDesc pr[6] = {
        {5, 3, 30, Ea, Et, o_w[0], nullptr},
        {5, 4, 40, Ea, Ev, o_w[1], o_Ea},
        {3, 5, 50, Et, Ea, o_w[2], nullptr},
        {3, 4, 60, Et, Ev, o_w[3], o_Et},
        {4, 5, 70, Ev, Ea, o_w[4], nullptr},
        {4, 3, 80, Ev, Et, o_w[5], o_Ev},
    };
    const int acg = (int)(B * 4 / 256);
    for (int i = 0; i < 6; ++i) {
        PairDesc& p = pr[i];
        int Mq = (int)(B * p.Tq), Mk = (int)(B * p.Tk);
        proj_gemm_kernel<<<Mq / 32, 64, 0, stream>>>(p.Eq, P(p.pb + 0), P(p.pb + 1), Qb);
        proj_gemm_kernel<<<Mk / 32, 64, 0, stream>>>(p.Ekv, P(p.pb + 2), P(p.pb + 3), Kb);
        proj_gemm_kernel<<<Mk / 32, 64, 0, stream>>>(p.Ekv, P(p.pb + 4), P(p.pb + 5), Vb);
        switch (i) {
            case 0: attn_core_kernel<5, 3><<<acg, 256, 0, stream>>>(Qb, Kb, Vb, Ab, p.wout); break;
            case 1: attn_core_kernel<5, 4><<<acg, 256, 0, stream>>>(Qb, Kb, Vb, Ab, p.wout); break;
            case 2: attn_core_kernel<3, 5><<<acg, 256, 0, stream>>>(Qb, Kb, Vb, Ab, p.wout); break;
            case 3: attn_core_kernel<3, 4><<<acg, 256, 0, stream>>>(Qb, Kb, Vb, Ab, p.wout); break;
            case 4: attn_core_kernel<4, 5><<<acg, 256, 0, stream>>>(Qb, Kb, Vb, Ab, p.wout); break;
            case 5: attn_core_kernel<4, 3><<<acg, 256, 0, stream>>>(Qb, Kb, Vb, Ab, p.wout); break;
        }
        oproj_ln_kernel<<<Mq / 32, 64, 0, stream>>>(
            Ab, p.Eq, P(p.pb + 6), P(p.pb + 7), P(p.pb + 8), P(p.pb + 9),
            p.Eq, p.enc);
    }

    // ---- classifier ----
    classifier_kernel<<<(int)(B / 32), 64, 0, stream>>>(
        Ea, Et, Ev, P(90), P(91), P(92), P(93), o_logits);
}
